// DeepGATBlockV2_70265664962912
// MI455X (gfx1250) — matrix operand order repaired and checked by reading, not compiled, not run on hardware
//
#include <hip/hip_runtime.h>

// Problem constants from the reference
constexpr int DIM   = 128;   // hidden D (== GEMM K for all three GEMMs)
constexpr int HEADS = 4;     // H
#define NEG_SLOPE 0.2f
#define RMS_EPS   1.1920929e-07f

typedef __attribute__((ext_vector_type(16))) __bf16 v16bf;
typedef __attribute__((ext_vector_type(8)))  __bf16 v8bf;
typedef __attribute__((ext_vector_type(4)))  __bf16 v4bf;
typedef __attribute__((ext_vector_type(8)))  float  v8f;
typedef __attribute__((ext_vector_type(4)))  float  f32x4;

// ---------------------------------------------------------------------------
// WMMA GEMM:  C[M,Nout] = A[M,128] * B[128,Nout]  (+bias)(+relu)
// fp32 -> bf16 staged in LDS, f32 accumulate via v_wmma_f32_16x16x32_bf16.
// Block = 256 threads = 8 wave32; block tile 128(M) x 64(N); wave tile 32x32.
// A staged row-major (pitch 40 bf16 = 80B: 16B-aligned 8-elt chunks, 20-bank
// row stride -> 16 rows land in 16 distinct banks). B staged TRANSPOSED
// (Bt[n][k]) so each lane's fragment is two 16-byte chunks at k = 8*hl and 16 + 8*hl (2x ds_load_b128).
// ---------------------------------------------------------------------------
#define TM 128
#define TN 64
#define TK 32
#define KDIM 128
#define APITCH 40
#define BPITCH 40

__global__ __launch_bounds__(256) void gemm_bf16_wmma(
    const float* __restrict__ A, const float* __restrict__ B,
    const float* __restrict__ bias, float* __restrict__ C,
    int M, int Nout, int flags)
{
  __shared__ __attribute__((aligned(16))) __bf16 As[TM * APITCH];
  __shared__ __attribute__((aligned(16))) __bf16 Bt[TN * BPITCH];

  const int tid  = threadIdx.x;
  const int wave = tid >> 5;
  const int lane = tid & 31;
  const int hl   = lane >> 4;   // wave half (0: lanes0-15, 1: lanes16-31)
  const int l16  = lane & 15;
  const int wm   = wave >> 1;   // 0..3  (M sub-tile)
  const int wn   = wave & 1;    // 0..1  (N sub-tile)
  const int row0 = blockIdx.y * TM;
  const int col0 = blockIdx.x * TN;

  // B staging coords: thread covers n = tid&63, k-chunk of 8 at (tid>>6)*8
  const int bn = tid & 63;
  const int bk = (tid >> 6) * 8;

  v8f acc[2][2];
  acc[0][0] = {}; acc[0][1] = {}; acc[1][0] = {}; acc[1][1] = {};

  #pragma unroll
  for (int kk = 0; kk < KDIM; kk += TK) {
    // ---- Stage A tile (128x32): branchless, vectorized b128 loads ----
    #pragma unroll
    for (int c = 0; c < 4; ++c) {
      int linear = tid + c * 256;            // 1024 float4 chunks
      int r    = linear >> 3;                // 8 chunks per 32-col row
      int col4 = (linear & 7) * 4;
      int gr   = row0 + r;
      int grc  = gr < M ? gr : M - 1;        // clamp: always a valid address
      float msk = gr < M ? 1.0f : 0.0f;
      f32x4 av = *(const f32x4*)&A[(size_t)grc * KDIM + kk + col4];
      av *= msk;
      *(v4bf*)&As[r * APITCH + col4] = __builtin_convertvector(av, v4bf);
    }
    // ---- Stage B tile (32k x 64n) transposed into Bt[n][k] ----
    {
      v8bf bv;
      #pragma unroll
      for (int j = 0; j < 8; ++j)
        bv[j] = (__bf16)B[(size_t)(kk + bk + j) * Nout + col0 + bn];
      *(v8bf*)&Bt[bn * BPITCH + bk] = bv;
    }
    if (kk + TK < KDIM)
      __builtin_prefetch(&A[(size_t)(row0 + (tid >> 1)) * KDIM + kk + TK], 0, 1);
    __syncthreads();

    // ---- Fragments: contiguous 16B chunks, assembled by shufflevector ----
    v16bf afrag[2], bfrag[2];
    #pragma unroll
    for (int mi = 0; mi < 2; ++mi) {
      int m = wm * 32 + mi * 16 + l16;
      // A 16x32 layout: half hl holds K {hl*8..hl*8+7, 16+hl*8..16+hl*8+7}
      v8bf lo = *(const v8bf*)&As[m * APITCH + hl * 8];
      v8bf hi = *(const v8bf*)&As[m * APITCH + 16 + hl * 8];
      afrag[mi] = __builtin_shufflevector(lo, hi, 0, 1, 2, 3, 4, 5, 6, 7,
                                          8, 9, 10, 11, 12, 13, 14, 15);
    }
    #pragma unroll
    for (int ni = 0; ni < 2; ++ni) {
      int n = wn * 32 + ni * 16 + l16;
      // B 32x16 layout: half hl holds K {hl*8..hl*8+7, 16+hl*8..16+hl*8+7}
      v8bf lo = *(const v8bf*)&Bt[n * BPITCH + hl * 8];
      v8bf hi = *(const v8bf*)&Bt[n * BPITCH + 16 + hl * 8];
      bfrag[ni] = __builtin_shufflevector(lo, hi, 0, 1, 2, 3, 4, 5, 6, 7,
                                          8, 9, 10, 11, 12, 13, 14, 15);
    }

    #pragma unroll
    for (int mi = 0; mi < 2; ++mi)
      #pragma unroll
      for (int ni = 0; ni < 2; ++ni)
        acc[mi][ni] = __builtin_amdgcn_wmma_f32_16x16x32_bf16(
            false, afrag[mi], false, bfrag[ni], (short)0, acc[mi][ni],
            false, false);
    __syncthreads();
  }

  // Epilogue: C/D layout -> VGPR r, lanes0-15: M=r ; lanes16-31: M=r+8
  const bool use_bias = flags & 1;
  const bool use_relu = flags & 2;
  #pragma unroll
  for (int mi = 0; mi < 2; ++mi) {
    #pragma unroll
    for (int ni = 0; ni < 2; ++ni) {
      int gn = col0 + wn * 32 + ni * 16 + l16;
      float bv = use_bias ? bias[gn] : 0.0f;
      #pragma unroll
      for (int r = 0; r < 8; ++r) {
        int gm = row0 + wm * 32 + mi * 16 + r + 8 * hl;
        if (gm < M) {
          float v = acc[mi][ni][r] + bv;
          if (use_relu) v = v > 0.0f ? v : 0.0f;
          C[(size_t)gm * Nout + gn] = v;
        }
      }
    }
  }
}

// ---------------------------------------------------------------------------
// Monotonic uint encoding of float for atomic max (handles negatives)
// ---------------------------------------------------------------------------
__device__ inline unsigned enc_f32(float f) {
  unsigned u = __float_as_uint(f);
  return (u & 0x80000000u) ? ~u : (u | 0x80000000u);
}
__device__ inline float dec_f32(unsigned u) {
  return (u & 0x80000000u) ? __uint_as_float(u & 0x7FFFFFFFu)
                           : __uint_as_float(~u);
}

// Zero agg / denom, init segment-max encodings
__global__ void init_scratch(float* __restrict__ agg, size_t nagg,
                             unsigned* __restrict__ m_enc,
                             float* __restrict__ denom, int nm)
{
  size_t i = (size_t)blockIdx.x * blockDim.x + threadIdx.x;
  size_t stride = (size_t)gridDim.x * blockDim.x;
  for (size_t j = i; j < nagg; j += stride) agg[j] = 0.0f;
  for (size_t j = i; j < (size_t)nm; j += stride) { m_enc[j] = 0u; denom[j] = 0.0f; }
}

// a_s[n,h] = <xp[n,h,:], att_src[h,:]>, same for a_d
__global__ void att_scores(const float* __restrict__ xp,
                           const float* __restrict__ att_s,
                           const float* __restrict__ att_d,
                           float* __restrict__ a_s, float* __restrict__ a_d,
                           int Nn)
{
  int i = blockIdx.x * blockDim.x + threadIdx.x;
  if (i >= Nn * HEADS) return;
  int n = i / HEADS, h = i % HEADS;
  const float* v  = xp + ((size_t)n * HEADS + h) * DIM;
  const float* ws = att_s + h * DIM;
  const float* wd = att_d + h * DIM;
  float s = 0.0f, t = 0.0f;
  #pragma unroll 4
  for (int d = 0; d < DIM; ++d) { float xv = v[d]; s += xv * ws[d]; t += xv * wd[d]; }
  a_s[i] = s; a_d[i] = t;
}

__device__ inline void edge_nodes(const int* __restrict__ ei, int E_, int idx,
                                  int& s, int& d) {
  if (idx < E_) { s = ei[idx]; d = ei[E_ + idx]; }
  else          { s = d = idx - E_; }     // appended self loops
}

__device__ inline float leaky(float v) { return v > 0.0f ? v : NEG_SLOPE * v; }

// Pass 1: segment max over incoming edges (atomic max on encoded uint)
__global__ void edge_max(const int* __restrict__ ei,
                         const float* __restrict__ a_s,
                         const float* __restrict__ a_d,
                         unsigned* __restrict__ m_enc, int Etot, int E_)
{
  int i = blockIdx.x * blockDim.x + threadIdx.x;
  if (i >= Etot * HEADS) return;
  int e = i >> 2, h = i & 3;
  int s, d; edge_nodes(ei, E_, e, s, d);
  float v = leaky(a_s[s * HEADS + h] + a_d[d * HEADS + h]);
  atomicMax(&m_enc[d * HEADS + h], enc_f32(v));
}

// Pass 2: segment sum of exp(e - m)
__global__ void edge_sum(const int* __restrict__ ei,
                         const float* __restrict__ a_s,
                         const float* __restrict__ a_d,
                         const unsigned* __restrict__ m_enc,
                         float* __restrict__ denom, int Etot, int E_)
{
  int i = blockIdx.x * blockDim.x + threadIdx.x;
  if (i >= Etot * HEADS) return;
  int e = i >> 2, h = i & 3;
  int s, d; edge_nodes(ei, E_, e, s, d);
  float v = leaky(a_s[s * HEADS + h] + a_d[d * HEADS + h]);
  float mv = dec_f32(m_enc[d * HEADS + h]);
  atomicAdd(&denom[d * HEADS + h], __expf(v - mv));
}

// Pass 3: agg[dst,h,:] += alpha * xp[src,h,:]  — one wave32 per (edge,head)
__global__ __launch_bounds__(256) void edge_agg(
    const int* __restrict__ ei,
    const float* __restrict__ a_s, const float* __restrict__ a_d,
    const unsigned* __restrict__ m_enc, const float* __restrict__ denom,
    const float* __restrict__ xp, float* __restrict__ agg, int Etot, int E_)
{
  int wid  = blockIdx.x * 8 + (threadIdx.x >> 5);
  int lane = threadIdx.x & 31;
  if (wid >= Etot * HEADS) return;
  int e = wid >> 2, h = wid & 3;
  int s, d; edge_nodes(ei, E_, e, s, d);
  float v  = leaky(a_s[s * HEADS + h] + a_d[d * HEADS + h]);
  float mv = dec_f32(m_enc[d * HEADS + h]);
  float alpha = __expf(v - mv) / denom[d * HEADS + h];
  const float* xs = xp  + ((size_t)s * HEADS + h) * DIM;
  float*       ag = agg + ((size_t)d * HEADS + h) * DIM;
  #pragma unroll
  for (int j = 0; j < DIM / 32; ++j) {
    int dd = lane + 32 * j;
    atomicAdd(&ag[dd], alpha * xs[dd]);
  }
}

// Post-GAT: x = rmsnorm(x + mean_h(agg) + bias, w) — one wave32 per node
__global__ __launch_bounds__(256) void gat_mean_rmsnorm(
    const float* __restrict__ xin, const float* __restrict__ agg,
    const float* __restrict__ bias, const float* __restrict__ w,
    float* __restrict__ xout, int Nn)
{
  int node = blockIdx.x * 8 + (threadIdx.x >> 5);
  int lane = threadIdx.x & 31;
  if (node >= Nn) return;
  float y[DIM / 32];
  float ss = 0.0f;
  #pragma unroll
  for (int j = 0; j < DIM / 32; ++j) {
    int dd = lane + 32 * j;
    const float* ag = agg + (size_t)node * HEADS * DIM + dd;
    float att = 0.25f * (ag[0] + ag[DIM] + ag[2 * DIM] + ag[3 * DIM]) + bias[dd];
    float v = xin[(size_t)node * DIM + dd] + att;
    y[j] = v; ss += v * v;
  }
  #pragma unroll
  for (int o = 16; o > 0; o >>= 1) ss += __shfl_xor(ss, o, 32);
  float r = rsqrtf(ss * (1.0f / DIM) + RMS_EPS);
  #pragma unroll
  for (int j = 0; j < DIM / 32; ++j) {
    int dd = lane + 32 * j;
    xout[(size_t)node * DIM + dd] = y[j] * r * w[dd];
  }
}

// Post-FFN: out = rmsnorm(xin + add, w)
__global__ __launch_bounds__(256) void resid_rmsnorm(
    const float* __restrict__ xin, const float* __restrict__ add,
    const float* __restrict__ w, float* __restrict__ xout, int Nn)
{
  int node = blockIdx.x * 8 + (threadIdx.x >> 5);
  int lane = threadIdx.x & 31;
  if (node >= Nn) return;
  float y[DIM / 32];
  float ss = 0.0f;
  #pragma unroll
  for (int j = 0; j < DIM / 32; ++j) {
    int dd = lane + 32 * j;
    float v = xin[(size_t)node * DIM + dd] + add[(size_t)node * DIM + dd];
    y[j] = v; ss += v * v;
  }
  #pragma unroll
  for (int o = 16; o > 0; o >>= 1) ss += __shfl_xor(ss, o, 32);
  float r = rsqrtf(ss * (1.0f / DIM) + RMS_EPS);
  #pragma unroll
  for (int j = 0; j < DIM / 32; ++j) {
    int dd = lane + 32 * j;
    xout[(size_t)node * DIM + dd] = y[j] * r * w[dd];
  }
}

// ---------------------------------------------------------------------------
extern "C" void kernel_launch(void* const* d_in, const int* in_sizes, int n_in,
                              void* d_out, int out_size, void* d_ws, size_t ws_size,
                              hipStream_t stream)
{
  const int Nn = in_sizes[0] / DIM;          // 20000
  const int Ll = in_sizes[4] / DIM;          // 3 (bias_gat = L*D)
  const int Ee = in_sizes[11] / 2;           // 320000
  const int Etot = Ee + Nn;                  // + self loops
  const int HD = HEADS * DIM;                // 512

  const float* x_in    = (const float*)d_in[0];
  const float* W_gat   = (const float*)d_in[1];
  const float* att_src = (const float*)d_in[2];
  const float* att_dst = (const float*)d_in[3];
  const float* bias_g  = (const float*)d_in[4];
  const float* W1      = (const float*)d_in[5];
  const float* b1      = (const float*)d_in[6];
  const float* W2      = (const float*)d_in[7];
  const float* b2      = (const float*)d_in[8];
  const float* n1w     = (const float*)d_in[9];
  const float* n2w     = (const float*)d_in[10];
  const int*   ei      = (const int*)d_in[11];

  // Workspace carving (fp32)
  float* ws    = (float*)d_ws;
  float* xcur  = ws;                                   // N*D
  float* xp    = xcur + (size_t)Nn * DIM;              // N*H*D
  float* agg   = xp   + (size_t)Nn * HD;               // N*H*D
  float* a_s   = agg  + (size_t)Nn * HD;               // N*H
  float* a_d   = a_s  + (size_t)Nn * HEADS;            // N*H
  float* denom = a_d  + (size_t)Nn * HEADS;            // N*H
  unsigned* m_enc = (unsigned*)(denom + (size_t)Nn * HEADS); // N*H
  float* h1    = (float*)(m_enc + (size_t)Nn * HEADS); // N*D
  float* h2    = h1 + (size_t)Nn * DIM;                // N*D

  hipMemcpyAsync(xcur, x_in, sizeof(float) * (size_t)Nn * DIM,
                 hipMemcpyDeviceToDevice, stream);

  const int EH = Etot * HEADS;
  const dim3 blk(256);

  for (int l = 0; l < Ll; ++l) {
    const float* Wg_l  = W_gat   + (size_t)l * DIM * HD;
    const float* as_l  = att_src + (size_t)l * HD;
    const float* ad_l  = att_dst + (size_t)l * HD;
    const float* bg_l  = bias_g  + (size_t)l * DIM;
    const float* W1_l  = W1      + (size_t)l * DIM * DIM;
    const float* b1_l  = b1      + (size_t)l * DIM;
    const float* W2_l  = W2      + (size_t)l * DIM * DIM;
    const float* b2_l  = b2      + (size_t)l * DIM;
    const float* n1_l  = n1w     + (size_t)l * DIM;
    const float* n2_l  = n2w     + (size_t)l * DIM;

    // xp = x @ W_gat
    gemm_bf16_wmma<<<dim3(HD / TN, (Nn + TM - 1) / TM), blk, 0, stream>>>(
        xcur, Wg_l, nullptr, xp, Nn, HD, 0);

    init_scratch<<<2048, blk, 0, stream>>>(agg, (size_t)Nn * HD, m_enc, denom,
                                           Nn * HEADS);
    att_scores<<<(Nn * HEADS + 255) / 256, blk, 0, stream>>>(xp, as_l, ad_l,
                                                             a_s, a_d, Nn);
    edge_max<<<(EH + 255) / 256, blk, 0, stream>>>(ei, a_s, a_d, m_enc, Etot, Ee);
    edge_sum<<<(EH + 255) / 256, blk, 0, stream>>>(ei, a_s, a_d, m_enc, denom,
                                                   Etot, Ee);
    edge_agg<<<(EH + 7) / 8, blk, 0, stream>>>(ei, a_s, a_d, m_enc, denom, xp,
                                               agg, Etot, Ee);
    // x = rmsnorm(x + mean_h(agg) + bias_gat, norm1_w)
    gat_mean_rmsnorm<<<(Nn + 7) / 8, blk, 0, stream>>>(xcur, agg, bg_l, n1_l,
                                                       xcur, Nn);
    // FFN
    gemm_bf16_wmma<<<dim3(DIM / TN, (Nn + TM - 1) / TM), blk, 0, stream>>>(
        xcur, W1_l, b1_l, h1, Nn, DIM, /*bias|relu*/ 3);
    gemm_bf16_wmma<<<dim3(DIM / TN, (Nn + TM - 1) / TM), blk, 0, stream>>>(
        h1, W2_l, b2_l, h2, Nn, DIM, /*bias*/ 1);

    float* out_l = (l == Ll - 1) ? (float*)d_out : xcur;
    resid_rmsnorm<<<(Nn + 7) / 8, blk, 0, stream>>>(xcur, h2, n2_l, out_l, Nn);
  }
}
